// LSTMHistoryEmbedder_30021821399483
// MI455X (gfx1250) — matrix operand order repaired and checked by reading, not compiled, not run on hardware
//
#include <hip/hip_runtime.h>

// ---------------- problem constants ----------------
#define SE       1024          // hidden / input size
#define T_STEPS  512
#define BATCH    32
#define G4       4096          // 4*SE gate rows
#define KCAT     2048          // SE (input) + SE (recurrent)
#define NBLK     64            // persistent grid: 64 blocks x 256 thr = 512 waves
#define NTHR     256
#define GSZ      (NBLK * NTHR) // 16384 threads

typedef __attribute__((ext_vector_type(16))) __bf16   v16bf;
typedef __attribute__((ext_vector_type(8)))  float    v8f;
typedef __attribute__((ext_vector_type(4)))  unsigned u32x4;

union FragBF { v16bf v; u32x4 u[2]; };

__device__ __forceinline__ unsigned short f2bf(float f) {
  unsigned u = __float_as_uint(f);
  u += 0x7fffu + ((u >> 16) & 1u);           // round-to-nearest-even
  return (unsigned short)(u >> 16);
}

// ---------------- conversion kernels ----------------
__global__ void cvt_weights(const float* __restrict__ Wih, const float* __restrict__ Whh,
                            unsigned short* __restrict__ wcat, size_t n) {
  size_t idx = (size_t)blockIdx.x * blockDim.x + threadIdx.x;
  if (idx >= n) return;
  size_t l   = idx / ((size_t)G4 * KCAT);
  size_t rem = idx % ((size_t)G4 * KCAT);
  size_t row = rem / KCAT;
  size_t k   = rem % KCAT;
  float v = (k < SE) ? Wih[(l * G4 + row) * SE + k]
                     : Whh[(l * G4 + row) * SE + (k - SE)];
  wcat[idx] = f2bf(v);
}

__global__ void cvt_bias(const float* __restrict__ bih, const float* __restrict__ bhh,
                         float* __restrict__ bsum, size_t n) {
  size_t idx = (size_t)blockIdx.x * blockDim.x + threadIdx.x;
  if (idx >= n) return;
  bsum[idx] = bih[idx] + bhh[idx];
}

__global__ void cvt_x(const float* __restrict__ x, unsigned short* __restrict__ xb, size_t n) {
  size_t idx = (size_t)blockIdx.x * blockDim.x + threadIdx.x;
  if (idx >= n) return;
  xb[idx] = f2bf(x[idx]);
}

// ---------------- grid-wide barrier (monotonic counter, memset once per launch) ----
__device__ __forceinline__ void gsync(unsigned* cnt, unsigned& barCount) {
  __threadfence();                 // release: make my gates/h stores visible at device scope
  __syncthreads();
  if (threadIdx.x == 0) {
    atomicAdd(cnt, 1u);
    ++barCount;
    const unsigned target = barCount * (unsigned)NBLK;
    while (__hip_atomic_load(cnt, __ATOMIC_ACQUIRE, __HIP_MEMORY_SCOPE_AGENT) < target)
      __builtin_amdgcn_s_sleep(2);
  }
  __syncthreads();
  __threadfence();                 // acquire: invalidate stale near-cache lines
}

// ---------------- persistent per-layer LSTM kernel ----------------
// Per step t: gates[32][4096] = [x_t ; h] @ Wcat^T + b   (bf16 WMMA, f32 acc)
// then elementwise i,f,g,o -> c,h ; h also stored bf16 as next step's A and as
// next layer's input sequence.
__global__ void __launch_bounds__(NTHR)
lstm_layer(const unsigned short* __restrict__ xin,   // [32][512][1024] bf16
           const unsigned short* __restrict__ wcat,  // [4096][2048] bf16 (this layer)
           const float* __restrict__ bsum,           // [4096]
           unsigned short* __restrict__ xout,        // [32][512][1024] bf16 (layer output)
           unsigned short* __restrict__ hbf,         // [32][1024] bf16
           float* __restrict__ cbuf,                 // [32][1024] f32
           float* __restrict__ gates,                // [32][4096] f32 scratch
           float* hf32,                              // [32][1024] f32 (d_out on last layer) or null
           unsigned* counter, unsigned barBase)
{
  const int tid  = threadIdx.x;
  const int gtid = blockIdx.x * NTHR + tid;
  const int lane = tid & 31;
  const int wid  = blockIdx.x * (NTHR / 32) + (tid >> 5);  // 0..511
  const int mtile = wid >> 8;                              // 0..1   (16 batch rows each)
  const int ntile = wid & 255;                             // 0..255 (16 gate cols each)
  const int mbase = mtile * 16;
  const int nbase = ntile * 16;
  const int lm = lane & 15;
  const int kh = lane >> 4;

  unsigned barCount = barBase;

  // zero-init h, c for this layer
  for (int e = gtid; e < BATCH * SE; e += GSZ) { cbuf[e] = 0.f; hbf[e] = 0; }
  gsync(counter, barCount);

  // B fragment: lane lm owns gate-row (nbase+lm) of Wcat; its 16 K-values are k0+8kh+0..7 and k0+16+8kh+0..7.
  const unsigned short* Brow = wcat + (size_t)(nbase + lm) * KCAT;
  const float bias = bsum[nbase + lm];
  const unsigned short* Ah = hbf + (size_t)(mbase + lm) * SE;

  for (int t = 0; t < T_STEPS; ++t) {
    const unsigned short* Ax = xin + ((size_t)(mbase + lm) * T_STEPS + t) * SE;
    v8f acc = {};

    // K = 0..1023 : input contribution x_t @ Wih^T
    #pragma unroll 4
    for (int k0 = 0; k0 < SE; k0 += 32) {
      FragBF a, b;
      a.u[0] = *reinterpret_cast<const u32x4*>(Ax + k0 + (kh << 3));
      a.u[1] = *reinterpret_cast<const u32x4*>(Ax + k0 + 16 + (kh << 3));
      b.u[0] = *reinterpret_cast<const u32x4*>(Brow + k0 + (kh << 3));
      b.u[1] = *reinterpret_cast<const u32x4*>(Brow + k0 + 16 + (kh << 3));
      acc = __builtin_amdgcn_wmma_f32_16x16x32_bf16(false, a.v, false, b.v,
                                                    (short)0, acc, false, false);
    }
    // K = 1024..2047 : recurrent contribution h_{t-1} @ Whh^T
    #pragma unroll 4
    for (int k0 = 0; k0 < SE; k0 += 32) {
      FragBF a, b;
      a.u[0] = *reinterpret_cast<const u32x4*>(Ah + k0 + (kh << 3));
      a.u[1] = *reinterpret_cast<const u32x4*>(Ah + k0 + 16 + (kh << 3));
      b.u[0] = *reinterpret_cast<const u32x4*>(Brow + SE + k0 + (kh << 3));
      b.u[1] = *reinterpret_cast<const u32x4*>(Brow + SE + k0 + 16 + (kh << 3));
      acc = __builtin_amdgcn_wmma_f32_16x16x32_bf16(false, a.v, false, b.v,
                                                    (short)0, acc, false, false);
    }

    // scatter gate pre-activations: lane = col nbase+lm, VGPR r = row r + 8*kh
    #pragma unroll
    for (int r = 0; r < 8; ++r) {
      const int mrow = mbase + r + (kh << 3);
      gates[(size_t)mrow * G4 + nbase + lm] = acc[r] + bias;
    }
    gsync(counter, barCount);

    // elementwise gate math (i,f,g,o are 1024 columns apart -> needs full gates row)
    for (int e = gtid; e < BATCH * SE; e += GSZ) {
      const int m = e >> 10;
      const int j = e & (SE - 1);
      const float* gm = gates + (size_t)m * G4;
      const float gi = gm[j];
      const float gf = gm[j + SE];
      const float gg = gm[j + 2 * SE];
      const float go = gm[j + 3 * SE];
      const float ig = 1.f / (1.f + __expf(-gi));
      const float fg = 1.f / (1.f + __expf(-gf));
      const float gc = tanhf(gg);
      const float og = 1.f / (1.f + __expf(-go));
      const float cN = fg * cbuf[e] + ig * gc;
      const float h  = og * tanhf(cN);
      cbuf[e] = cN;
      const unsigned short hb = f2bf(h);
      hbf[e] = hb;
      xout[((size_t)m * T_STEPS + t) * SE + j] = hb;
      if (hf32) hf32[e] = h;     // last layer: final step leaves the answer in d_out
    }
    gsync(counter, barCount);
  }
}

// ---------------- host launcher ----------------
extern "C" void kernel_launch(void* const* d_in, const int* in_sizes, int n_in,
                              void* d_out, int out_size, void* d_ws, size_t ws_size,
                              hipStream_t stream) {
  const float* x   = (const float*)d_in[0];
  const float* Wih = (const float*)d_in[1];
  const float* Whh = (const float*)d_in[2];
  const float* bih = (const float*)d_in[3];
  const float* bhh = (const float*)d_in[4];
  float* out = (float*)d_out;
  char* ws = (char*)d_ws;

  size_t off = 0;
  auto alloc = [&](size_t bytes) {
    size_t p = off;
    off += (bytes + 255) & ~(size_t)255;
    return p;
  };
  unsigned*       counter = (unsigned*)      (ws + alloc(64));
  float*          bsum    = (float*)         (ws + alloc(3ull * G4 * sizeof(float)));
  float*          cbuf    = (float*)         (ws + alloc((size_t)BATCH * SE * sizeof(float)));
  unsigned short* hbf     = (unsigned short*)(ws + alloc((size_t)BATCH * SE * sizeof(unsigned short)));
  float*          gatesb  = (float*)         (ws + alloc((size_t)BATCH * G4 * sizeof(float)));
  unsigned short* wcat    = (unsigned short*)(ws + alloc(3ull * G4 * KCAT * sizeof(unsigned short)));
  unsigned short* x0      = (unsigned short*)(ws + alloc((size_t)BATCH * T_STEPS * SE * sizeof(unsigned short)));
  unsigned short* x1      = (unsigned short*)(ws + alloc((size_t)BATCH * T_STEPS * SE * sizeof(unsigned short)));
  // total ws use ~118 MB

  hipMemsetAsync(counter, 0, 64, stream);   // graph-capture-legal memset node

  { size_t n = 3ull * G4 * KCAT;
    cvt_weights<<<(unsigned)((n + 255) / 256), 256, 0, stream>>>(Wih, Whh, wcat, n); }
  { size_t n = 3ull * G4;
    cvt_bias<<<(unsigned)((n + 255) / 256), 256, 0, stream>>>(bih, bhh, bsum, n); }
  { size_t n = (size_t)BATCH * T_STEPS * SE;
    cvt_x<<<(unsigned)((n + 255) / 256), 256, 0, stream>>>(x, x0, n); }

  unsigned short* xin = x0;
  unsigned short* xo  = x1;
  for (int l = 0; l < 3; ++l) {
    lstm_layer<<<NBLK, NTHR, 0, stream>>>(
        xin, wcat + (size_t)l * G4 * KCAT, bsum + (size_t)l * G4,
        xo, hbf, cbuf, gatesb, (l == 2) ? out : nullptr,
        counter, (unsigned)(l * (1 + 2 * T_STEPS)));
    unsigned short* tmp = xin; xin = xo; xo = tmp;
  }

  (void)in_sizes; (void)n_in; (void)out_size; (void)ws_size;
}
